// DownBlock_72164040508063
// MI455X (gfx1250) — compile-verified, matrix operand order checked by reading, not run on hardware
//
#include <hip/hip_runtime.h>

typedef _Float16 h8  __attribute__((ext_vector_type(8)));
typedef _Float16 h16 __attribute__((ext_vector_type(16)));
typedef float    f8  __attribute__((ext_vector_type(8)));

#define HH    128
#define WW    128
#define CIN   64
#define COUT  128
#define FIN   576            // CIN*9
#define KTOT  5184           // FIN*9 (silu + 8 basis per feature)
#define KC    288            // chunk = 32 features * 9 = 9 WMMA k-steps
#define NCHUNK 18
#define SA    296            // padded LDS stride (halfs)
#define SB    296
#define SC    132            // C-stage stride (floats)
#define NPIX  65536

// ---- cubic B-spline basis on uniform knots g_i = 0.4*i - 2.2, i=0..11 ----
__device__ __forceinline__ void bspline8(float v, float out[8]) {
    float b0[11];
#pragma unroll
    for (int i = 0; i < 11; ++i) {
        float gi = 0.4f * i - 2.2f;
        b0[i] = (v >= gi && v < gi + 0.4f) ? 1.0f : 0.0f;
    }
    float b1[10];
#pragma unroll
    for (int i = 0; i < 10; ++i) {
        float gi = 0.4f * i - 2.2f;
        b1[i] = ((v - gi) * b0[i] + ((gi + 0.8f) - v) * b0[i + 1]) * 2.5f;
    }
    float b2[9];
#pragma unroll
    for (int i = 0; i < 9; ++i) {
        float gi = 0.4f * i - 2.2f;
        b2[i] = ((v - gi) * b1[i] + ((gi + 1.2f) - v) * b1[i + 1]) * 1.25f;
    }
#pragma unroll
    for (int i = 0; i < 8; ++i) {
        float gi = 0.4f * i - 2.2f;
        out[i] = ((v - gi) * b2[i] + ((gi + 1.6f) - v) * b2[i + 1]) * (1.0f / 1.2f);
    }
}

__device__ __forceinline__ float silu_f(float v) { return v / (1.0f + __expf(-v)); }

// A/B fragment load matching CDNA5 16-bit 16x32 operand layout:
// lane%16 -> row, VGPR0..3 -> k = 8*(lane/16)..+7, VGPR4..7 -> k+16..+23
__device__ __forceinline__ h16 load_frag(const _Float16* base, int r0, int stride,
                                         int kb, int lane) {
    const _Float16* p = base + (r0 + (lane & 15)) * stride + kb + ((lane >> 4) << 3);
    h8 lo = *(const h8*)p;
    h8 hi = *(const h8*)(p + 16);
    h16 r;
#pragma unroll
    for (int i = 0; i < 8; ++i) { r[i] = lo[i]; r[i + 8] = hi[i]; }
    return r;
}

// ---- kernel 0: pack weights to f16 [COUT][KTOT], zero BN stat accumulators ----
extern "C" __global__ __launch_bounds__(256)
void kan_prep(const float* __restrict__ bw, const float* __restrict__ sw,
              _Float16* __restrict__ wc, float* __restrict__ stats) {
    int idx = blockIdx.x * 256 + threadIdx.x;
    if (idx < 512) stats[idx] = 0.0f;
    if (idx >= COUT * KTOT) return;
    int o = idx / KTOT;
    int k = idx - o * KTOT;
    int f = k / 9;
    int j = k - f * 9;
    float v = (j == 0) ? bw[o * FIN + f] : sw[(o * FIN + f) * 8 + (j - 1)];
    wc[idx] = (_Float16)v;
}

// ---- kernel 1: fused im2col + basis-gen + WMMA GEMM. Writes raw y [B,C,H,W] ----
extern "C" __global__ __launch_bounds__(256)
void kan_gemm(const float* __restrict__ x, const _Float16* __restrict__ wc,
              float* __restrict__ yraw) {
    extern __shared__ char smem[];
    _Float16* At = (_Float16*)smem;          // 128 x SA halfs
    _Float16* Bt = At + 128 * SA;            // 128 x SB halfs
    float*    Cs = (float*)smem;             // reused after K-loop: 128 x SC floats

    const int tid  = threadIdx.x;
    const int lane = tid & 31;
    const int wave = tid >> 5;               // 0..7
    const int wm   = wave & 3;               // 4 groups of 32 rows
    const int wn   = wave >> 1 >> 1;         // wave>>2: 2 groups of 64 cols
    const int row  = blockIdx.x;             // b*128 + y
    const int b    = row >> 7;
    const int yb   = row & 127;
    const float* xb = x + b * CIN * HH * WW;

    f8 acc[2][4];
#pragma unroll
    for (int i = 0; i < 2; ++i)
#pragma unroll
        for (int j = 0; j < 4; ++j)
#pragma unroll
            for (int r = 0; r < 8; ++r) acc[i][j][r] = 0.0f;

    for (int c = 0; c < NCHUNK; ++c) {
        // --- generate A tile: 128 pixels x 32 features x 9 columns (f16) ---
#pragma unroll 2
        for (int it = 0; it < 16; ++it) {
            int task = it * 256 + tid;       // 0..4095
            int xx = task & 127;
            int fl = task >> 7;              // 0..31
            int f  = c * 32 + fl;
            int cin = f / 9;
            int t9  = f - cin * 9;
            int dy  = t9 / 3 - 1;
            int dx  = t9 - (t9 / 3) * 3 - 1;
            int yy  = yb + dy;
            int xc  = xx + dx;
            float v = 0.0f;
            if ((unsigned)yy < (unsigned)HH && (unsigned)xc < (unsigned)WW)
                v = xb[(cin * HH + yy) * WW + xc];
            float bas[8];
            bspline8(v, bas);
            _Float16* dst = At + xx * SA + fl * 9;
            dst[0] = (_Float16)silu_f(v);
#pragma unroll
            for (int j = 0; j < 8; ++j) dst[1 + j] = (_Float16)bas[j];
        }
        // --- stream B chunk (N-major) into LDS ---
#pragma unroll 2
        for (int it = 0; it < 18; ++it) {
            int idx = it * 256 + tid;        // 0..4607 = 128 rows * 36 octets
            int r   = idx / 36;
            int c8  = idx - r * 36;
            *(h8*)(Bt + r * SB + c8 * 8) =
                *(const h8*)(wc + r * KTOT + c * KC + c8 * 8);
        }
        __syncthreads();

        // --- 9 k-steps of v_wmma_f32_16x16x32_f16 per wave ---
#pragma unroll
        for (int ks = 0; ks < 9; ++ks) {
            int kb = ks * 32;
            h16 af[2], bf[4];
            af[0] = load_frag(At, wm * 32 + 0,  SA, kb, lane);
            af[1] = load_frag(At, wm * 32 + 16, SA, kb, lane);
#pragma unroll
            for (int nf = 0; nf < 4; ++nf)
                bf[nf] = load_frag(Bt, wn * 64 + nf * 16, SB, kb, lane);
#pragma unroll
            for (int mi = 0; mi < 2; ++mi)
#pragma unroll
                for (int nf = 0; nf < 4; ++nf)
                    acc[mi][nf] = __builtin_amdgcn_wmma_f32_16x16x32_f16(
                        false, af[mi], false, bf[nf], (short)0, acc[mi][nf],
                        false, false);
        }
        __syncthreads();
    }

    // --- transpose C through LDS, coalesced store of raw y in [B,C,H,W] ---
    {
        int h  = (lane >> 4) & 1;
        int nl = lane & 15;
#pragma unroll
        for (int mi = 0; mi < 2; ++mi)
#pragma unroll
            for (int nf = 0; nf < 4; ++nf) {
                int m0 = wm * 32 + mi * 16 + h * 8;
                int n0 = wn * 64 + nf * 16 + nl;
#pragma unroll
                for (int r = 0; r < 8; ++r)
                    Cs[(m0 + r) * SC + n0] = acc[mi][nf][r];
            }
    }
    __syncthreads();
#pragma unroll 4
    for (int it = 0; it < 64; ++it) {
        int idx = it * 256 + tid;            // 0..16383
        int o = idx >> 7;
        int r = idx & 127;
        yraw[((b * COUT + o) * HH + yb) * WW + r] = Cs[r * SC + o];
    }
}

// ---- kernel 2: per-channel sum / sumsq over raw y ----
extern "C" __global__ __launch_bounds__(256)
void bn_stats(const float* __restrict__ yraw, float* __restrict__ sum,
              float* __restrict__ sumsq) {
    __shared__ float rs[256], rq[256];
    int plane = blockIdx.x;                  // b*128 + o, 512 planes
    int o = plane & 127;
    const float* p = yraw + plane * (HH * WW);
    float s = 0.0f, q = 0.0f;
#pragma unroll 4
    for (int i = 0; i < 64; ++i) {
        float v = p[i * 256 + threadIdx.x];
        s += v; q += v * v;
    }
    rs[threadIdx.x] = s; rq[threadIdx.x] = q;
    __syncthreads();
    for (int st = 128; st > 0; st >>= 1) {
        if (threadIdx.x < st) {
            rs[threadIdx.x] += rs[threadIdx.x + st];
            rq[threadIdx.x] += rq[threadIdx.x + st];
        }
        __syncthreads();
    }
    if (threadIdx.x == 0) {
        atomicAdd(&sum[o], rs[0]);
        atomicAdd(&sumsq[o], rq[0]);
    }
}

// ---- kernel 3: finalize BN scale/bias ----
extern "C" __global__ void bn_final(const float* __restrict__ sum,
                                    const float* __restrict__ sumsq,
                                    const float* __restrict__ gamma,
                                    const float* __restrict__ beta,
                                    float* __restrict__ scale,
                                    float* __restrict__ bias) {
    int o = threadIdx.x;
    const float inv = 1.0f / (float)NPIX;
    float m = sum[o] * inv;
    float v = sumsq[o] * inv - m * m;
    float sc = gamma[o] * rsqrtf(v + 1e-5f);
    scale[o] = sc;
    bias[o] = beta[o] - m * sc;
}

// ---- kernel 4: fused BN + SiLU + 2x2 maxpool (thread owns one 2x2 quad) ----
extern "C" __global__ __launch_bounds__(256)
void bn_silu_pool(float* __restrict__ skip, float* __restrict__ pooled,
                  const float* __restrict__ scale, const float* __restrict__ bias) {
    int idx = blockIdx.x * 256 + threadIdx.x;    // 4*128*64*64 total
    int pw = idx & 63;
    int ph = (idx >> 6) & 63;
    int o  = (idx >> 12) & 127;
    int b  = idx >> 19;
    float sc = scale[o], bi = bias[o];
    int base = ((b * COUT + o) * HH + 2 * ph) * WW + 2 * pw;
    float v00 = skip[base],      v01 = skip[base + 1];
    float v10 = skip[base + WW], v11 = skip[base + WW + 1];
    float a00 = silu_f(v00 * sc + bi);
    float a01 = silu_f(v01 * sc + bi);
    float a10 = silu_f(v10 * sc + bi);
    float a11 = silu_f(v11 * sc + bi);
    skip[base]          = a00;
    skip[base + 1]      = a01;
    skip[base + WW]     = a10;
    skip[base + WW + 1] = a11;
    float mx = fmaxf(fmaxf(a00, a01), fmaxf(a10, a11));
    pooled[idx] = mx;
}

extern "C" void kernel_launch(void* const* d_in, const int* in_sizes, int n_in,
                              void* d_out, int out_size, void* d_ws, size_t ws_size,
                              hipStream_t stream) {
    const float* x     = (const float*)d_in[0];
    const float* bw    = (const float*)d_in[1];
    const float* sw    = (const float*)d_in[2];
    const float* gamma = (const float*)d_in[3];
    const float* beta  = (const float*)d_in[4];

    float* pooled = (float*)d_out;                       // [4,128,64,64]
    float* skip   = pooled + 4 * COUT * 64 * 64;         // [4,128,128,128]

    _Float16* wc = (_Float16*)d_ws;                      // 128*5184 f16 = 1.33MB
    float* stats = (float*)((char*)d_ws + (size_t)COUT * KTOT * sizeof(_Float16));
    float* sum   = stats;
    float* sumsq = stats + 128;
    float* scale = stats + 256;
    float* bias  = stats + 384;

    kan_prep<<<(COUT * KTOT + 255) / 256, 256, 0, stream>>>(bw, sw, wc, stats);

    size_t smem = (size_t)2 * 128 * SA * sizeof(_Float16);   // 151,552 B of 320KB WGP LDS
    kan_gemm<<<512, 256, smem, stream>>>(x, wc, skip);

    bn_stats<<<512, 256, 0, stream>>>(skip, sum, sumsq);
    bn_final<<<1, 128, 0, stream>>>(sum, sumsq, gamma, beta, scale, bias);
    bn_silu_pool<<<4 * COUT * 64 * 64 / 256, 256, 0, stream>>>(skip, pooled, scale, bias);
}
